// RNN_9715216024222
// MI455X (gfx1250) — compile-verified, matrix operand order checked by reading, not run on hardware
//
#include <hip/hip_runtime.h>
#include <hip/hip_bf16.h>
#include <math.h>

// RNN LM: T=128, B=64, E=H=1024, L=2, V=10000
#define T_ 128
#define B_ 64
#define E_ 1024
#define H_ 1024
#define V_ 10000
#define KREC 2048  // E + H

typedef __attribute__((ext_vector_type(16))) __bf16 v16bf;
typedef __attribute__((ext_vector_type(8)))  float  v8f;

// ---------------- small utility kernels ----------------

__global__ void cast_f32_bf16(const float* __restrict__ src, __bf16* __restrict__ dst, int n) {
  int i = blockIdx.x * blockDim.x + threadIdx.x;
  if (i < n) dst[i] = (__bf16)src[i];
}

// xseq[(t*B+b)*E + e] = bf16(emb[tok[t*B+b]*E + e])
__global__ void embed_gather(const int* __restrict__ tok, const float* __restrict__ emb,
                             __bf16* __restrict__ xseq, int n) {
  int i = blockIdx.x * blockDim.x + threadIdx.x;
  if (i >= n) return;
  int row = i >> 10;       // / E_
  int e   = i & (E_ - 1);
  int idx = tok[row];
  xseq[i] = (__bf16)emb[(size_t)idx * E_ + e];
}

// hidden (L,B,H) fp32 -> bf16 ping buffers (slot 0)
__global__ void init_hidden(const float* __restrict__ hid, __bf16* __restrict__ h0,
                            __bf16* __restrict__ h1, int n /* B*H */) {
  int i = blockIdx.x * blockDim.x + threadIdx.x;
  if (i < n) {
    h0[i] = (__bf16)hid[i];
    h1[i] = (__bf16)hid[n + i];
  }
}

// ---------------- recurrent layer GEMM ----------------
// out[b][h] = tanh( sum_k [Ax|Ah][b][k] * W[h][k] + bias[h] )
// Per wave: 16(M) x 64(N) tile, K-loop step 32, 4 WMMAs reusing the A fragment.
// Grid: 8 blocks x 256 threads = 64 waves = (B/16) * (H/64). Latency-bound, keep simple.
__global__ __launch_bounds__(256)
void rnn_layer_gemm(const __bf16* __restrict__ Ax, const __bf16* __restrict__ Ah,
                    const __bf16* __restrict__ W, const float* __restrict__ bias,
                    __bf16* __restrict__ hOut, __bf16* __restrict__ topOut,
                    float* __restrict__ hFinal) {
  int wid  = blockIdx.x * (blockDim.x >> 5) + (threadIdx.x >> 5);
  int lane = threadIdx.x & 31;
  int half = lane >> 4;
  int r    = lane & 15;

  int mBase = (wid & 3) * 16;   // batch tile
  int nBase = (wid >> 2) * 64;  // hidden chunk (4 tiles of 16)

  v8f acc[4] = {v8f{}, v8f{}, v8f{}, v8f{}};

  const int arow = mBase + r;
  for (int k0 = 0; k0 < KREC; k0 += 32) {
    int kk = k0 + half * 16;
    // concat select: first E_ columns come from Ax, rest from Ah (tiles never straddle)
    const __bf16* ap = (kk < E_) ? (Ax + (size_t)arow * E_ + kk)
                                 : (Ah + (size_t)arow * H_ + (kk - E_));
    v16bf a = *(const v16bf*)ap;
    __builtin_prefetch(ap + 32, 0, 0);

    const __bf16* wp = W + (size_t)(nBase + r) * KREC + kk;
    v16bf b0 = *(const v16bf*)(wp);
    v16bf b1 = *(const v16bf*)(wp + (size_t)16 * KREC);
    v16bf b2 = *(const v16bf*)(wp + (size_t)32 * KREC);
    v16bf b3 = *(const v16bf*)(wp + (size_t)48 * KREC);
    __builtin_prefetch(wp + 32, 0, 0);

    acc[0] = __builtin_amdgcn_wmma_f32_16x16x32_bf16(false, a, false, b0, (short)0, acc[0], false, false);
    acc[1] = __builtin_amdgcn_wmma_f32_16x16x32_bf16(false, a, false, b1, (short)0, acc[1], false, false);
    acc[2] = __builtin_amdgcn_wmma_f32_16x16x32_bf16(false, a, false, b2, (short)0, acc[2], false, false);
    acc[3] = __builtin_amdgcn_wmma_f32_16x16x32_bf16(false, a, false, b3, (short)0, acc[3], false, false);
  }

  // Epilogue: D element i -> (M = i + 8*half, N = lane&15)
  #pragma unroll
  for (int j = 0; j < 4; j++) {
    int gn = nBase + j * 16 + r;
    float bb = bias[gn];
    #pragma unroll
    for (int i = 0; i < 8; i++) {
      int gm = mBase + i + 8 * half;
      float v = tanhf(acc[j][i] + bb);
      hOut[(size_t)gm * H_ + gn] = (__bf16)v;
      if (topOut) topOut[(size_t)gm * H_ + gn] = (__bf16)v;
      if (hFinal) hFinal[(size_t)gm * H_ + gn] = v;
    }
  }
}

// ---------------- output projection GEMM ----------------
// logits[m][v] = sum_h top[m][h] * Wout[v][h] + bout[v];  M = 8192, N = 10000, K = 1024
// Block = 8 waves covering a 128-row M super-tile, all sharing one 64-col N chunk.
// B (Wout) chunk is staged into LDS via async global->LDS (double-buffered, ASYNCcnt),
// giving 8x reuse of B per LDS fill. LDS rows padded to 40 elems (80B): 16B-aligned
// rows and conflict-free 16-lane fragment reads (stride 20 banks mod 64).
#define LDS_STRIDE 40
__global__ __launch_bounds__(256)
void out_gemm(const __bf16* __restrict__ top, const __bf16* __restrict__ Wout,
              const float* __restrict__ bout, float* __restrict__ logits) {
  __shared__ __bf16 bstage[2][64 * LDS_STRIDE];

  int tid    = threadIdx.x;
  int waveid = tid >> 5;
  int lane   = tid & 31;
  int half   = lane >> 4;
  int r      = lane & 15;

  int mBase = (blockIdx.y * 8 + waveid) * 16;  // 64 super-tiles * 8 waves -> 8192 rows
  int nBase = blockIdx.x * 64;                 // 157 chunks (last partially valid)

  // staging map: thread -> (row n = tid/4 of 64, 16B segment seg = tid%4 of the 32-K slab)
  int sn  = tid >> 2;
  int seg = tid & 3;

  auto stage = [&](int buf, int k0) {
    int wrow = nBase + sn;
    if (wrow >= V_) wrow = V_ - 1;  // clamp OOB rows (stores are guarded in epilogue)
    const __bf16* g = Wout + (size_t)wrow * H_ + k0 + seg * 8;
    unsigned lds = (unsigned)(uintptr_t)(&bstage[buf][sn * LDS_STRIDE + seg * 8]);
    asm volatile("global_load_async_to_lds_b128 %0, %1, off"
                 :: "v"(lds), "v"((unsigned long long)(uintptr_t)g)
                 : "memory");
  };

  v8f acc[4] = {v8f{}, v8f{}, v8f{}, v8f{}};

  stage(0, 0);  // prime buffer 0

  const __bf16* arowp = top + (size_t)(mBase + r) * H_;
  for (int k0 = 0; k0 < H_; k0 += 32) {
    int cur = (k0 >> 5) & 1;
    if (k0 + 32 < H_) {
      stage(cur ^ 1, k0 + 32);                       // fill other buffer ahead
      asm volatile("s_wait_asynccnt 0x1" ::: "memory");  // my writes to `cur` done (in-order)
    } else {
      asm volatile("s_wait_asynccnt 0x0" ::: "memory");
    }
    __syncthreads();  // all waves' writes to `cur` visible

    int kk = k0 + half * 16;
    v16bf a = *(const v16bf*)(arowp + kk);
    __builtin_prefetch(arowp + kk + 32, 0, 0);

    #pragma unroll
    for (int j = 0; j < 4; j++) {
      const __bf16* bp = &bstage[cur][(j * 16 + r) * LDS_STRIDE + half * 16];
      v16bf b = *(const v16bf*)bp;  // 2x ds_load_b128, conflict-free
      acc[j] = __builtin_amdgcn_wmma_f32_16x16x32_bf16(false, a, false, b, (short)0, acc[j], false, false);
    }
    __syncthreads();  // everyone done reading `cur` before it is re-staged
  }

  #pragma unroll
  for (int j = 0; j < 4; j++) {
    int tileBase = nBase + j * 16;
    if (tileBase + 16 > V_) continue;  // V_ % 16 == 0 -> valid tiles fully in range
    int gn = tileBase + r;
    float bb = bout[gn];
    #pragma unroll
    for (int i = 0; i < 8; i++) {
      int gm = mBase + i + 8 * half;
      logits[(size_t)gm * V_ + gn] = acc[j][i] + bb;
    }
  }
}

// ---------------- launcher ----------------

extern "C" void kernel_launch(void* const* d_in, const int* in_sizes, int n_in,
                              void* d_out, int out_size, void* d_ws, size_t ws_size,
                              hipStream_t stream) {
  const int*   tok  = (const int*)d_in[0];
  const float* hid  = (const float*)d_in[1];
  const float* emb  = (const float*)d_in[2];
  const float* W0   = (const float*)d_in[3];
  const float* b0   = (const float*)d_in[4];
  const float* Wn   = (const float*)d_in[5];
  const float* bn   = (const float*)d_in[6];
  const float* Wout = (const float*)d_in[7];
  const float* bout = (const float*)d_in[8];

  float* logits = (float*)d_out;                  // (T,B,V)
  float* hfinal = logits + (size_t)T_ * B_ * V_;  // (L,B,H)

  char* ws = (char*)d_ws;
  __bf16* W0b   = (__bf16*)ws; ws += (size_t)H_ * KREC * 2;
  __bf16* Wnb   = (__bf16*)ws; ws += (size_t)H_ * KREC * 2;
  __bf16* Woutb = (__bf16*)ws; ws += (size_t)V_ * H_ * 2;
  __bf16* xseq  = (__bf16*)ws; ws += (size_t)T_ * B_ * E_ * 2;
  __bf16* topb  = (__bf16*)ws; ws += (size_t)T_ * B_ * H_ * 2;
  __bf16* h0buf[2];
  h0buf[0] = (__bf16*)ws; ws += (size_t)B_ * H_ * 2;
  h0buf[1] = (__bf16*)ws; ws += (size_t)B_ * H_ * 2;
  __bf16* h1buf[2];
  h1buf[0] = (__bf16*)ws; ws += (size_t)B_ * H_ * 2;
  h1buf[1] = (__bf16*)ws; ws += (size_t)B_ * H_ * 2;

  int n;
  n = H_ * KREC; cast_f32_bf16<<<(n + 255) / 256, 256, 0, stream>>>(W0, W0b, n);
  n = H_ * KREC; cast_f32_bf16<<<(n + 255) / 256, 256, 0, stream>>>(Wn, Wnb, n);
  n = V_ * H_;   cast_f32_bf16<<<(n + 255) / 256, 256, 0, stream>>>(Wout, Woutb, n);

  n = B_ * H_;
  init_hidden<<<(n + 255) / 256, 256, 0, stream>>>(hid, h0buf[0], h1buf[0], n);

  n = T_ * B_ * E_;
  embed_gather<<<(n + 255) / 256, 256, 0, stream>>>(tok, emb, xseq, n);

  // sequential recurrence: ping-pong hidden buffers by t parity
  for (int t = 0; t < T_; t++) {
    int rp = t & 1, wp = rp ^ 1;
    bool last = (t == T_ - 1);
    rnn_layer_gemm<<<8, 256, 0, stream>>>(
        xseq + (size_t)t * B_ * E_, h0buf[rp], W0b, b0,
        h0buf[wp], (__bf16*)nullptr, last ? hfinal : (float*)nullptr);
    rnn_layer_gemm<<<8, 256, 0, stream>>>(
        h0buf[wp], h1buf[rp], Wnb, bn,
        h1buf[wp], topb + (size_t)t * B_ * H_,
        last ? (hfinal + (size_t)B_ * H_) : (float*)nullptr);
  }

  // output projection: 157 N-chunks x 64 M super-tiles, 8 waves/block
  dim3 grid(157, 64);
  out_gemm<<<grid, 256, 0, stream>>>(topb, Woutb, bout, logits);
}
